// ExchangeRateModel_56856777064540
// MI455X (gfx1250) — matrix operand order repaired and checked by reading, not compiled, not run on hardware
//
#include <hip/hip_runtime.h>

// ---------------------------------------------------------------------------
// LSTM (H=50) over [B=2048, T=1024, D=1] -> last hidden -> lstm2 cell -> fc.
// Strategy: 16-batch-row tiles, bf16 WMMA 16x16x32, K padded 50(+x col)->64,
// gates padded 4x50 -> 4x64 = 256 cols. 4 waves/block, one gate block each.
// Activations use v_exp_f32 + v_rcp_f32 (no IEEE divide expansion).
// ---------------------------------------------------------------------------

typedef __attribute__((ext_vector_type(16))) __bf16 v16bf;
typedef __attribute__((ext_vector_type(8)))  float  v8f;
typedef __attribute__((ext_vector_type(4)))  float  vf4;

#define HH   50      // hidden size
#define BT   16      // batch tile (M)
#define HSTR 80      // h LDS row stride (ushorts), 16B-aligned, bank-skewed

union ABits { v16bf v; vf4 f4[2]; };   // one WMMA 16x32 bf16 operand (8 VGPRs)
union CBits { v8f  v; vf4 f4[2]; };    // one WMMA 16x16 f32 accum (8 VGPRs)

static __device__ __forceinline__ unsigned short f2bf(float f) {
  unsigned u = __float_as_uint(f);
  unsigned r = (u + 0x7FFFu + ((u >> 16) & 1u)) >> 16;   // round-to-nearest-even
  return (unsigned short)r;
}
static __device__ __forceinline__ float frcp(float x) {
  return __builtin_amdgcn_rcpf(x);              // v_rcp_f32, no NR expansion
}
static __device__ __forceinline__ float sigmoidf_(float x) {
  return frcp(1.0f + __expf(-x));               // v_exp + v_rcp
}
static __device__ __forceinline__ float tanhf_(float x) {
  return 1.0f - 2.0f * frcp(__expf(2.0f * x) + 1.0f);   // v_exp + v_rcp + fma
}

// ---------------------------------------------------------------------------
// Pack B operands (bf16, WMMA B-fragment layout) into d_ws:
//   buf 0: lstm1  B[K=64][N=256], row K=50 carries Wih1 (x folded into GEMM)
//   buf 1: lstm2  B[K=64][N=256] from Wih2 (zero recurrent state => Whh2 unused)
// Fragment = (N-tile t, K-chunk c): 32 lanes x 16 bf16, lane-linear (1KB).
// Element e of lane l within chunk c:  K = c*32 + (e&7) + (e>>3)*16 + (l<16 ? 0 : 8),
//                                      N = t*16 + (l&15).
// ---------------------------------------------------------------------------
__global__ void pack_weights(const float* __restrict__ Wih1,
                             const float* __restrict__ Whh1,
                             const float* __restrict__ Wih2,
                             unsigned short* __restrict__ pack) {
  int idx = blockIdx.x * blockDim.x + threadIdx.x;   // 0 .. 65535
  int buf    = idx >> 15;
  int r      = idx & 32767;
  int frag   = r >> 9;            // t*2 + c
  int within = r & 511;
  int lane   = within >> 4;
  int e      = within & 15;
  int t = frag >> 1;
  int c = frag & 1;
  int k = c * 32 + (e & 7) + ((e >> 3) << 4) + ((lane < 16) ? 0 : 8);
  int n = t * 16 + (lane & 15);
  int gate = n >> 6;
  int j    = n & 63;
  float val = 0.0f;
  if (j < HH) {
    int no = gate * HH + j;
    if (buf == 0) {
      if (k < HH)       val = Whh1[no * HH + k];
      else if (k == HH) val = Wih1[no];          // appended x column
    } else {
      if (k < HH)       val = Wih2[no * HH + k];
    }
  }
  pack[idx] = f2bf(val);
}

// ---------------------------------------------------------------------------
// Main kernel: one block = 16 batch rows, 4 waves (wave w owns gate block w
// for the GEMM, and hidden columns [16w,16w+16) for the c/h update).
// ---------------------------------------------------------------------------
__global__ void __launch_bounds__(128)
lstm_wmma(const float* __restrict__ x,
          const float* __restrict__ bih1, const float* __restrict__ bhh1,
          const float* __restrict__ bih2, const float* __restrict__ bhh2,
          const float* __restrict__ fcW,  const float* __restrict__ fcb,
          const unsigned short* __restrict__ pack,
          float* __restrict__ out, int T) {
  __shared__ unsigned short h_lds[BT * HSTR];  // h_t (bf16), row-major, col 50 = x_t
  __shared__ float g_lds[16 * 256];            // 16 gate fragments (f32), 16KB

  const int tid  = threadIdx.x;
  const int w    = tid >> 5;          // wave id == gate block (i,f,g,o)
  const int l    = tid & 31;
  const int ln   = l & 15;
  const int half = (l < 16) ? 0 : 8;  // A K-offset / C row-offset for hi lanes
  const int bbase = blockIdx.x * BT;
  const int mycol = w * 16 + ln;      // hidden column this lane updates

  // ---- init h state: zeros, x_0 in column 50 -------------------------------
  for (int i = tid; i < BT * HSTR; i += blockDim.x) h_lds[i] = 0;
  __syncthreads();
  if (tid < BT) h_lds[tid * HSTR + HH] = f2bf(x[(bbase + tid) * T]);

  // ---- per-lane constants --------------------------------------------------
  float bias1[4], bias2[4];
  #pragma unroll
  for (int tt = 0; tt < 4; ++tt) {
    int jj = tt * 16 + ln;
    bias1[tt] = (jj < HH) ? (bih1[w * HH + jj] + bhh1[w * HH + jj]) : 0.0f;
    bias2[tt] = (jj < HH) ? (bih2[w * HH + jj] + bhh2[w * HH + jj]) : 0.0f;
  }
  const float wcol = (mycol < HH) ? fcW[mycol] : 0.0f;

  // ---- resident lstm1 B fragments: 4 N-tiles x 2 K-chunks (64 VGPRs) -------
  v16bf Bf[4][2];
  #pragma unroll
  for (int tt = 0; tt < 4; ++tt)
    #pragma unroll
    for (int c = 0; c < 2; ++c) {
      const vf4* p = (const vf4*)(pack + ((w * 4 + tt) * 2 + c) * 512 + l * 16);
      ABits b; b.f4[0] = p[0]; b.f4[1] = p[1];
      Bf[tt][c] = b.v;
    }

  v8f cstate = {};                    // cell state slice lives in VGPRs
  __syncthreads();

  // ---- sequential scan over T timesteps ------------------------------------
  #pragma unroll 1
  for (int t = 0; t < T; ++t) {
    __builtin_prefetch(x + (bbase + ln) * T + t + 8, 0, 1);  // global_prefetch_b8

    // A operands (h_t, K=64) straight from LDS in WMMA A layout
    ABits a0, a1;
    const unsigned short* hr = h_lds + ln * HSTR;
    a0.f4[0] = *(const vf4*)(hr + half);
    a0.f4[1] = *(const vf4*)(hr + 16 + half);
    a1.f4[0] = *(const vf4*)(hr + 32 + half);
    a1.f4[1] = *(const vf4*)(hr + 48 + half);

    // gates for my gate block: 4 output tiles, C init = bias splat
    #pragma unroll
    for (int tt = 0; tt < 4; ++tt) {
      v8f acc;
      #pragma unroll
      for (int i = 0; i < 8; ++i) acc[i] = bias1[tt];
      acc = __builtin_amdgcn_wmma_f32_16x16x32_bf16(false, a0.v, false, Bf[tt][0],
                                                    (short)0, acc, false, false);
      acc = __builtin_amdgcn_wmma_f32_16x16x32_bf16(false, a1.v, false, Bf[tt][1],
                                                    (short)0, acc, false, false);
      CBits cb; cb.v = acc;
      float* gp = g_lds + (w * 4 + tt) * 256 + l * 8;
      *(vf4*)(gp)     = cb.f4[0];
      *(vf4*)(gp + 4) = cb.f4[1];
    }
    __syncthreads();

    // gather i,f,g,o tiles covering my hidden-column slice
    CBits gi, gf, gg, go;
    {
      const float* p;
      p = g_lds + (0 * 4 + w) * 256 + l * 8; gi.f4[0] = *(const vf4*)p; gi.f4[1] = *(const vf4*)(p + 4);
      p = g_lds + (1 * 4 + w) * 256 + l * 8; gf.f4[0] = *(const vf4*)p; gf.f4[1] = *(const vf4*)(p + 4);
      p = g_lds + (2 * 4 + w) * 256 + l * 8; gg.f4[0] = *(const vf4*)p; gg.f4[1] = *(const vf4*)(p + 4);
      p = g_lds + (3 * 4 + w) * 256 + l * 8; go.f4[0] = *(const vf4*)p; go.f4[1] = *(const vf4*)(p + 4);
    }

    // c/h update; write h_{t+1} (bf16) back; column 50 gets x_{t+1}
    #pragma unroll
    for (int v = 0; v < 8; ++v) {
      float iv = sigmoidf_(gi.v[v]);
      float fv = sigmoidf_(gf.v[v]);
      float gv = tanhf_(gg.v[v]);
      float ov = sigmoidf_(go.v[v]);
      float cn = fv * cstate[v] + iv * gv;
      cstate[v] = cn;
      float hn = ov * tanhf_(cn);
      int row = v + half;
      unsigned short st;
      if (mycol == HH) {
        float xn = (t + 1 < T) ? x[(bbase + row) * T + (t + 1)] : 0.0f;
        st = f2bf(xn);
      } else {
        st = f2bf(hn);
      }
      h_lds[row * HSTR + mycol] = st;
    }
    __syncthreads();
  }

  // ---- lstm2: single cell from zero state; input = h1 (in h_lds, col50=0) --
  v16bf B2[4][2];
  #pragma unroll
  for (int tt = 0; tt < 4; ++tt)
    #pragma unroll
    for (int c = 0; c < 2; ++c) {
      const vf4* p = (const vf4*)(pack + 32768 + ((w * 4 + tt) * 2 + c) * 512 + l * 16);
      ABits b; b.f4[0] = p[0]; b.f4[1] = p[1];
      B2[tt][c] = b.v;
    }
  {
    ABits a0, a1;
    const unsigned short* hr = h_lds + ln * HSTR;
    a0.f4[0] = *(const vf4*)(hr + half);
    a0.f4[1] = *(const vf4*)(hr + 16 + half);
    a1.f4[0] = *(const vf4*)(hr + 32 + half);
    a1.f4[1] = *(const vf4*)(hr + 48 + half);
    #pragma unroll
    for (int tt = 0; tt < 4; ++tt) {
      v8f acc;
      #pragma unroll
      for (int i = 0; i < 8; ++i) acc[i] = bias2[tt];
      acc = __builtin_amdgcn_wmma_f32_16x16x32_bf16(false, a0.v, false, B2[tt][0],
                                                    (short)0, acc, false, false);
      acc = __builtin_amdgcn_wmma_f32_16x16x32_bf16(false, a1.v, false, B2[tt][1],
                                                    (short)0, acc, false, false);
      CBits cb; cb.v = acc;
      float* gp = g_lds + (w * 4 + tt) * 256 + l * 8;
      *(vf4*)(gp)     = cb.f4[0];
      *(vf4*)(gp + 4) = cb.f4[1];
    }
  }
  __syncthreads();

  // activations (c0 = 0 => f-gate irrelevant) + fc partial products
  {
    CBits gi, gg, go;
    const float* p;
    p = g_lds + (0 * 4 + w) * 256 + l * 8; gi.f4[0] = *(const vf4*)p; gi.f4[1] = *(const vf4*)(p + 4);
    p = g_lds + (2 * 4 + w) * 256 + l * 8; gg.f4[0] = *(const vf4*)p; gg.f4[1] = *(const vf4*)(p + 4);
    p = g_lds + (3 * 4 + w) * 256 + l * 8; go.f4[0] = *(const vf4*)p; go.f4[1] = *(const vf4*)(p + 4);
    __syncthreads();                       // g_lds reused as [16][64] f32 below
    #pragma unroll
    for (int v = 0; v < 8; ++v) {
      float iv = sigmoidf_(gi.v[v]);
      float gv = tanhf_(gg.v[v]);
      float ov = sigmoidf_(go.v[v]);
      float c2 = iv * gv;
      float h2 = ov * tanhf_(c2);
      int row = v + half;
      g_lds[row * 64 + mycol] = h2 * wcol;
    }
  }
  __syncthreads();

  // fc reduction: out[b] = sum_j h2[b][j]*fcW[j] + fcb
  if (tid < BT) {
    float s = fcb[0];
    for (int c = 0; c < 64; ++c) s += g_lds[tid * 64 + c];
    out[bbase + tid] = s;
  }
}

// ---------------------------------------------------------------------------
extern "C" void kernel_launch(void* const* d_in, const int* in_sizes, int n_in,
                              void* d_out, int out_size, void* d_ws, size_t ws_size,
                              hipStream_t stream) {
  const float* x    = (const float*)d_in[0];
  const float* Wih1 = (const float*)d_in[1];
  const float* Whh1 = (const float*)d_in[2];
  const float* bih1 = (const float*)d_in[3];
  const float* bhh1 = (const float*)d_in[4];
  const float* Wih2 = (const float*)d_in[5];
  // d_in[6] = Whh2: multiplied by the zero state in the reference -> unused
  const float* bih2 = (const float*)d_in[7];
  const float* bhh2 = (const float*)d_in[8];
  const float* fcW  = (const float*)d_in[9];
  const float* fcb  = (const float*)d_in[10];
  float* out = (float*)d_out;
  unsigned short* pack = (unsigned short*)d_ws;   // 128KB: two packed B buffers

  const int B = out_size;               // 2048
  const int T = in_sizes[0] / B;        // 1024

  pack_weights<<<dim3(256), dim3(256), 0, stream>>>(Wih1, Whh1, Wih2, pack);
  lstm_wmma<<<dim3(B / BT), dim3(128), 0, stream>>>(
      x, bih1, bhh1, bih2, bhh2, fcW, fcb, pack, out, T);
  (void)n_in; (void)ws_size;
}
